// MoEFFN_67645734912843
// MI455X (gfx1250) — compile-verified, matrix operand order checked by reading, not run on hardware
//
#include <hip/hip_runtime.h>
#include <hip/hip_bf16.h>
#include <math.h>

// MoE FFN for MI455X (gfx1250), wave32, bf16 WMMA with fp32 accumulation.
// Gathered top-2 routing: ~137 GFLOP of matmul, 268 MB fp32 weight stream
// (HBM floor ~11.5us @ 23.3 TB/s). fp32->bf16 conversion fused into GEMM
// tile loads via packed v_cvt_pk_bf16_f32; 128x64 tiles, ping-pong LDS
// double buffering (1 barrier/step), 4 WMMAs per wave per K-step.

#define DMODEL 1024
#define HDIM   4096
#define NEXP   8
#define TOPK   2
#define NTOK   4096   // B*T = 2*2048

typedef __attribute__((ext_vector_type(16))) __bf16 v16bf;
typedef __attribute__((ext_vector_type(2)))  __bf16 v2bf;
typedef __attribute__((ext_vector_type(8)))  float  v8f;
typedef __attribute__((ext_vector_type(2)))  float  v2f;

union AFrag { v16bf v; unsigned u[8]; };

// packed fp32->bf16 (RNE): <2 x float> -> <2 x bfloat> = one v_cvt_pk_bf16_f32
__device__ __forceinline__ unsigned pack2bf(float a, float b) {
    v2f f; f.x = a; f.y = b;
    union { v2bf h; unsigned u; } w;
    w.h = __builtin_convertvector(f, v2bf);
    return w.u;
}
__device__ __forceinline__ unsigned short f2bf(float a) {
    union { __bf16 h; unsigned short u; } w;
    w.h = (__bf16)a;
    return w.u;
}
__device__ __forceinline__ float gelu_erf(float h) {
    return 0.5f * h * (1.0f + erff(h * 0.70710678118654752f));
}

#define WMMA_BF16(A, B, C) \
    __builtin_amdgcn_wmma_f32_16x16x32_bf16(false, (A), false, (B), (short)0, (C), false, false)

// ---------------- utility kernels ----------------

__global__ void zero_f32(float* p, int n) {
    int i = blockIdx.x * blockDim.x + threadIdx.x;
    if (i < n) p[i] = 0.f;
}
__global__ void zero_i32(int* p, int n) {
    int i = blockIdx.x * blockDim.x + threadIdx.x;
    if (i < n) p[i] = 0;
}

// ---------------- router: one wave32 per token ----------------

__global__ void router_kernel(const float* __restrict__ x,
                              const unsigned char* __restrict__ mask,
                              const float* __restrict__ Wr,
                              int* __restrict__ counts,
                              int* __restrict__ topIdx,
                              float* __restrict__ topGate) {
    int wave = blockIdx.x * (blockDim.x >> 5) + (threadIdx.x >> 5);
    int lane = threadIdx.x & 31;
    if (wave >= NTOK) return;
    const float* xr = x + (size_t)wave * DMODEL;
    float acc[NEXP];
#pragma unroll
    for (int e = 0; e < NEXP; ++e) acc[e] = 0.f;
    for (int d = lane; d < DMODEL; d += 32) {
        float xv = xr[d];
#pragma unroll
        for (int e = 0; e < NEXP; ++e) acc[e] += xv * Wr[e * DMODEL + d];
    }
#pragma unroll
    for (int e = 0; e < NEXP; ++e)
        for (int o = 16; o > 0; o >>= 1)
            acc[e] += __shfl_xor(acc[e], o, 32);
    if (lane == 0) {
        float m = acc[0];
        for (int e = 1; e < NEXP; ++e) m = fmaxf(m, acc[e]);
        float p[NEXP]; float s = 0.f;
        for (int e = 0; e < NEXP; ++e) { p[e] = __expf(acc[e] - m); s += p[e]; }
        float inv = (mask[wave] ? 1.f : 0.f) / s;
        for (int e = 0; e < NEXP; ++e) p[e] *= inv;
        int i1 = 0; float v1 = p[0];
        for (int e = 1; e < NEXP; ++e) if (p[e] > v1) { v1 = p[e]; i1 = e; }
        int i2 = (i1 == 0) ? 1 : 0; float v2 = p[i2];
        for (int e = 0; e < NEXP; ++e)
            if (e != i1 && p[e] > v2) { v2 = p[e]; i2 = e; }
        float denom = v1 + v2 + 1e-9f;
        topIdx[wave * 2]     = i1;  topGate[wave * 2]     = v1 / denom;
        topIdx[wave * 2 + 1] = i2;  topGate[wave * 2 + 1] = v2 / denom;
        atomicAdd(&counts[i1], 1);
        atomicAdd(&counts[i2], 1);
    }
}

__global__ void scan_kernel(const int* __restrict__ counts, int* __restrict__ offsets) {
    if (threadIdx.x == 0 && blockIdx.x == 0) {
        int s = 0;
        for (int e = 0; e < NEXP; ++e) { offsets[e] = s; s += counts[e]; }
    }
}

__global__ void scatter_kernel(const int* __restrict__ topIdx,
                               const float* __restrict__ topGate,
                               const int* __restrict__ offsets,
                               int* __restrict__ cursors,
                               int* __restrict__ slotTok,
                               float* __restrict__ slotGate) {
    int t = blockIdx.x * blockDim.x + threadIdx.x;
    if (t >= NTOK) return;
#pragma unroll
    for (int k = 0; k < TOPK; ++k) {
        int e   = topIdx[t * TOPK + k];
        float g = topGate[t * TOPK + k];
        int pos = atomicAdd(&cursors[e], 1);
        int slot = offsets[e] + pos;
        slotTok[slot]  = t;
        slotGate[slot] = g;
    }
}

// ---------------- LayerNorm xhat: one wave32 per token ----------------

__global__ void xhat_kernel(const float* __restrict__ x, float* __restrict__ xhat) {
    int wave = blockIdx.x * (blockDim.x >> 5) + (threadIdx.x >> 5);
    int lane = threadIdx.x & 31;
    if (wave >= NTOK) return;
    const float* xr = x + (size_t)wave * DMODEL;
    float s = 0.f, s2 = 0.f;
    for (int d = lane; d < DMODEL; d += 32) { float v = xr[d]; s += v; s2 += v * v; }
    for (int o = 16; o > 0; o >>= 1) {
        s  += __shfl_xor(s,  o, 32);
        s2 += __shfl_xor(s2, o, 32);
    }
    float mean = s * (1.f / DMODEL);
    float var  = s2 * (1.f / DMODEL) - mean * mean;
    float rstd = rsqrtf(var + 1e-5f);
    float* xo = xhat + (size_t)wave * DMODEL;
    for (int d = lane; d < DMODEL; d += 32) xo[d] = (xr[d] - mean) * rstd;
}

// ---------------- b1eff = W1 @ ln_b + b1 ----------------

__global__ void b1eff_kernel(const float* __restrict__ W1,
                             const float* __restrict__ ln_b,
                             const float* __restrict__ b1,
                             float* __restrict__ b1eff) {
    int i = blockIdx.x * blockDim.x + threadIdx.x;
    if (i >= NEXP * HDIM) return;
    int e = i / HDIM;
    const float* wr = W1 + (size_t)i * DMODEL;
    const float* lb = ln_b + (size_t)e * DMODEL;
    float acc = 0.f;
    for (int d = 0; d < DMODEL; ++d) acc += wr[d] * lb[d];
    b1eff[i] = acc + b1[i];
}

// ---------------- GEMM1: Act = gelu((xhat*gamma_e) @ W1^T + b1eff), bf16 out --------
// 128x64 tile per 256-thread block, ping-pong LDS, 4 WMMAs/wave/K-step.

__global__ __launch_bounds__(256)
void gemm1_kernel(const float* __restrict__ xhat,
                  const float* __restrict__ W1,
                  const float* __restrict__ ln_g,
                  const float* __restrict__ b1eff,
                  const int* __restrict__ counts,
                  const int* __restrict__ offsets,
                  const int* __restrict__ slotTok,
                  unsigned short* __restrict__ Act) {
    const int e = blockIdx.z;
    const int count = counts[e];
    const int rowBase = blockIdx.y * 128;
    if (rowBase >= count) return;
    const int off = offsets[e];
    const int hBase = blockIdx.x * 64;
    const int rowsIn = min(128, count - rowBase);

    __shared__ unsigned As[2][128][16];   // 128 rows x 32 bf16
    __shared__ unsigned Bs[2][64][16];
    __shared__ int tokS[128];

    const int tid = threadIdx.x;
    if (tid < 128) tokS[tid] = (tid < rowsIn) ? slotTok[off + rowBase + tid] : 0;
    __syncthreads();

    // loader mapping: A -> 2 threads/row (16 elems), B -> 4 threads/row (8 elems)
    const int ar = tid >> 1, aseg = tid & 1;
    const int br = tid >> 2, bseg = tid & 3;
    const int tok = tokS[ar];
    const float* aSrc = xhat + (size_t)tok * DMODEL + aseg * 16;
    const float* gSrc = ln_g + (size_t)e * DMODEL + aseg * 16;
    const float* bSrc = W1 + ((size_t)e * HDIM + (hBase + br)) * DMODEL + bseg * 8;

    // compute mapping: wave owns M-subtiles {ms0,ms0+1} x N-subtiles {ns0,ns0+1}
    const int wid = tid >> 5, lane = tid & 31;
    const int g = lane >> 4, idx = lane & 15;
    const int ms0 = (wid & 3) * 2;
    const int ns0 = (wid >> 2) * 2;

    v8f acc00 = {}, acc01 = {}, acc10 = {}, acc11 = {};
    float aReg[16], bReg[8];

    // prologue: stage K-step 0 into LDS buffer 0
#pragma unroll
    for (int i = 0; i < 16; ++i) aReg[i] = aSrc[i] * gSrc[i];
#pragma unroll
    for (int i = 0; i < 8; ++i)  bReg[i] = bSrc[i];
#pragma unroll
    for (int i = 0; i < 8; ++i)  As[0][ar][aseg * 8 + i] = pack2bf(aReg[2 * i], aReg[2 * i + 1]);
#pragma unroll
    for (int i = 0; i < 4; ++i)  Bs[0][br][bseg * 4 + i] = pack2bf(bReg[2 * i], bReg[2 * i + 1]);
    __syncthreads();

    int buf = 0;
    for (int kc = 0; kc < DMODEL; kc += 32) {
        const int kn = kc + 32;
        if (kn < DMODEL) {   // issue next-step global loads early
#pragma unroll
            for (int i = 0; i < 16; ++i) aReg[i] = aSrc[kn + i] * gSrc[kn + i];
#pragma unroll
            for (int i = 0; i < 8; ++i)  bReg[i] = bSrc[kn + i];
        }
        AFrag fa0, fa1, fb0, fb1;
#pragma unroll
        for (int p = 0; p < 8; ++p) {
            int q = (p < 4) ? (p + 4 * g) : (p + 4 + 4 * g);
            fa0.u[p] = As[buf][ms0 * 16 + idx][q];
            fa1.u[p] = As[buf][ms0 * 16 + 16 + idx][q];
            fb0.u[p] = Bs[buf][ns0 * 16 + idx][q];
            fb1.u[p] = Bs[buf][ns0 * 16 + 16 + idx][q];
        }
        acc00 = WMMA_BF16(fa0.v, fb0.v, acc00);
        acc01 = WMMA_BF16(fa0.v, fb1.v, acc01);
        acc10 = WMMA_BF16(fa1.v, fb0.v, acc10);
        acc11 = WMMA_BF16(fa1.v, fb1.v, acc11);
        if (kn < DMODEL) {   // stage into alternate buffer
#pragma unroll
            for (int i = 0; i < 8; ++i)
                As[buf ^ 1][ar][aseg * 8 + i] = pack2bf(aReg[2 * i], aReg[2 * i + 1]);
#pragma unroll
            for (int i = 0; i < 4; ++i)
                Bs[buf ^ 1][br][bseg * 4 + i] = pack2bf(bReg[2 * i], bReg[2 * i + 1]);
        }
        __syncthreads();
        buf ^= 1;
    }

    const float* b1e = b1eff + (size_t)e * HDIM + hBase;
#pragma unroll
    for (int sm = 0; sm < 2; ++sm) {
        v8f* accRow0 = sm ? &acc10 : &acc00;
        v8f* accRow1 = sm ? &acc11 : &acc01;
#pragma unroll
        for (int v = 0; v < 8; ++v) {
            int m = (ms0 + sm) * 16 + v + 8 * g;
            if (m < rowsIn) {
                size_t rowG = (size_t)(off + rowBase + m) * HDIM + hBase;
                int n0 = ns0 * 16 + idx;
                Act[rowG + n0] = f2bf(gelu_erf((*accRow0)[v] + b1e[n0]));
                int n1 = n0 + 16;
                Act[rowG + n1] = f2bf(gelu_erf((*accRow1)[v] + b1e[n1]));
            }
        }
    }
}

// ---------------- GEMM2: out += gate * (Act @ W2^T + b2) ----------------

__global__ __launch_bounds__(256)
void gemm2_kernel(const unsigned short* __restrict__ Act,
                  const float* __restrict__ W2,
                  const float* __restrict__ b2,
                  const int* __restrict__ counts,
                  const int* __restrict__ offsets,
                  const int* __restrict__ slotTok,
                  const float* __restrict__ slotGate,
                  float* __restrict__ out) {
    const int e = blockIdx.z;
    const int count = counts[e];
    const int rowBase = blockIdx.y * 128;
    if (rowBase >= count) return;
    const int off = offsets[e];
    const int dBase = blockIdx.x * 64;
    const int rowsIn = min(128, count - rowBase);

    __shared__ unsigned As[2][128][16];
    __shared__ unsigned Bs[2][64][16];
    __shared__ int   tokS[128];
    __shared__ float gateS[128];

    const int tid = threadIdx.x;
    if (tid < 128) {
        if (tid < rowsIn) {
            tokS[tid]  = slotTok[off + rowBase + tid];
            gateS[tid] = slotGate[off + rowBase + tid];
        } else { tokS[tid] = 0; gateS[tid] = 0.f; }
    }
    __syncthreads();

    const int ar = tid >> 1, aseg = tid & 1;
    const int br = tid >> 2, bseg = tid & 3;
    const bool rValid = (ar < rowsIn);
    const unsigned* aU = (const unsigned*)(Act + (size_t)(off + rowBase + (rValid ? ar : 0)) * HDIM)
                         + aseg * 8;
    const float* bSrc = W2 + ((size_t)e * DMODEL + (dBase + br)) * HDIM + bseg * 8;

    const int wid = tid >> 5, lane = tid & 31;
    const int g = lane >> 4, idx = lane & 15;
    const int ms0 = (wid & 3) * 2;
    const int ns0 = (wid >> 2) * 2;

    v8f acc00 = {}, acc01 = {}, acc10 = {}, acc11 = {};
    unsigned aReg[8]; float bReg[8];

    // prologue: K-step 0
#pragma unroll
    for (int i = 0; i < 8; ++i) aReg[i] = rValid ? aU[i] : 0u;
#pragma unroll
    for (int i = 0; i < 8; ++i) bReg[i] = bSrc[i];
#pragma unroll
    for (int i = 0; i < 8; ++i) As[0][ar][aseg * 8 + i] = aReg[i];
#pragma unroll
    for (int i = 0; i < 4; ++i) Bs[0][br][bseg * 4 + i] = pack2bf(bReg[2 * i], bReg[2 * i + 1]);
    __syncthreads();

    int buf = 0;
    for (int kc = 0; kc < HDIM; kc += 32) {
        const int kn = kc + 32;
        if (kn < HDIM) {
#pragma unroll
            for (int i = 0; i < 8; ++i) aReg[i] = rValid ? aU[kn / 2 + i] : 0u;
#pragma unroll
            for (int i = 0; i < 8; ++i) bReg[i] = bSrc[kn + i];
        }
        AFrag fa0, fa1, fb0, fb1;
#pragma unroll
        for (int p = 0; p < 8; ++p) {
            int q = (p < 4) ? (p + 4 * g) : (p + 4 + 4 * g);
            fa0.u[p] = As[buf][ms0 * 16 + idx][q];
            fa1.u[p] = As[buf][ms0 * 16 + 16 + idx][q];
            fb0.u[p] = Bs[buf][ns0 * 16 + idx][q];
            fb1.u[p] = Bs[buf][ns0 * 16 + 16 + idx][q];
        }
        acc00 = WMMA_BF16(fa0.v, fb0.v, acc00);
        acc01 = WMMA_BF16(fa0.v, fb1.v, acc01);
        acc10 = WMMA_BF16(fa1.v, fb0.v, acc10);
        acc11 = WMMA_BF16(fa1.v, fb1.v, acc11);
        if (kn < HDIM) {
#pragma unroll
            for (int i = 0; i < 8; ++i) As[buf ^ 1][ar][aseg * 8 + i] = aReg[i];
#pragma unroll
            for (int i = 0; i < 4; ++i)
                Bs[buf ^ 1][br][bseg * 4 + i] = pack2bf(bReg[2 * i], bReg[2 * i + 1]);
        }
        __syncthreads();
        buf ^= 1;
    }

    const float* b2e = b2 + (size_t)e * DMODEL + dBase;
#pragma unroll
    for (int sm = 0; sm < 2; ++sm) {
        v8f* accRow0 = sm ? &acc10 : &acc00;
        v8f* accRow1 = sm ? &acc11 : &acc01;
#pragma unroll
        for (int v = 0; v < 8; ++v) {
            int m = (ms0 + sm) * 16 + v + 8 * g;
            if (m < rowsIn) {
                float gate = gateS[m];
                size_t orow = (size_t)tokS[m] * DMODEL + dBase;
                int n0 = ns0 * 16 + idx;
                atomicAdd(&out[orow + n0], ((*accRow0)[v] + b2e[n0]) * gate);
                int n1 = n0 + 16;
                atomicAdd(&out[orow + n1], ((*accRow1)[v] + b2e[n1]) * gate);
            }
        }
    }
}

// ---------------- host launcher ----------------

extern "C" void kernel_launch(void* const* d_in, const int* in_sizes, int n_in,
                              void* d_out, int out_size, void* d_ws, size_t ws_size,
                              hipStream_t stream) {
    const float* x            = (const float*)d_in[0];
    const unsigned char* mask = (const unsigned char*)d_in[1];
    const float* Wr           = (const float*)d_in[2];
    const float* ln_g         = (const float*)d_in[3];
    const float* ln_b         = (const float*)d_in[4];
    const float* W1           = (const float*)d_in[5];
    const float* b1           = (const float*)d_in[6];
    const float* W2           = (const float*)d_in[7];
    const float* b2           = (const float*)d_in[8];
    float* out = (float*)d_out;

    char* ws = (char*)d_ws;
    size_t off = 0;
    float* xhat   = (float*)(ws + off); off += (size_t)NTOK * DMODEL * 4;      // 16 MB
    float* b1eff  = (float*)(ws + off); off += (size_t)NEXP * HDIM * 4;        // 128 KB
    unsigned short* Act = (unsigned short*)(ws + off);
    off += (size_t)NTOK * TOPK * HDIM * 2;                                     // 64 MB
    int*   counts  = (int*)(ws + off);  off += NEXP * 4;
    int*   cursors = (int*)(ws + off);  off += NEXP * 4;   // contiguous with counts
    int*   offsets = (int*)(ws + off);  off += NEXP * 4;
    int*   topIdx  = (int*)(ws + off);  off += NTOK * TOPK * 4;
    float* topGate = (float*)(ws + off); off += NTOK * TOPK * 4;
    int*   slotTok = (int*)(ws + off);  off += NTOK * TOPK * 4;
    float* slotGate = (float*)(ws + off); off += NTOK * TOPK * 4;

    zero_f32<<<(NTOK * DMODEL + 255) / 256, 256, 0, stream>>>(out, NTOK * DMODEL);
    zero_i32<<<1, 64, 0, stream>>>(counts, 2 * NEXP);   // counts + cursors

    router_kernel<<<NTOK / 8, 256, 0, stream>>>(x, mask, Wr, counts, topIdx, topGate);
    scan_kernel<<<1, 1, 0, stream>>>(counts, offsets);
    scatter_kernel<<<(NTOK + 255) / 256, 256, 0, stream>>>(topIdx, topGate, offsets,
                                                           cursors, slotTok, slotGate);
    xhat_kernel<<<NTOK / 8, 256, 0, stream>>>(x, xhat);
    b1eff_kernel<<<(NEXP * HDIM + 255) / 256, 256, 0, stream>>>(W1, ln_b, b1, b1eff);

    dim3 g1(HDIM / 64, (NTOK + 127) / 128, NEXP);
    gemm1_kernel<<<g1, 256, 0, stream>>>(xhat, W1, ln_g, b1eff,
                                         counts, offsets, slotTok, Act);
    dim3 g2(DMODEL / 64, (NTOK + 127) / 128, NEXP);
    gemm2_kernel<<<g2, 256, 0, stream>>>(Act, W2, b2, counts, offsets,
                                         slotTok, slotGate, out);
}
